// ResidualBlock_63230508532142
// MI455X (gfx1250) — compile-verified, matrix operand order checked by reading, not run on hardware
//
#include <hip/hip_runtime.h>
#include <hip/hip_bf16.h>

// ---------------- problem constants ----------------
#define BB 8
#define HH 56
#define WW 56
#define MM (BB * HH * WW)   // 25088 spatial positions, divisible by 128
#define EPSI 1e-5f

typedef __attribute__((ext_vector_type(16))) __bf16 v16bf;
typedef __attribute__((ext_vector_type(8)))  float  v8f;
typedef __attribute__((ext_vector_type(4)))  unsigned int u32x4;

// =====================================================================
// Layout transposes: NCHW <-> NHWC (fp32)
// =====================================================================
__global__ __launch_bounds__(256)
void k_nchw2nhwc(const float* __restrict__ in, float* __restrict__ out, int C) {
  size_t i = (size_t)blockIdx.x * 256 + threadIdx.x;
  size_t total = (size_t)MM * C;
  if (i >= total) return;
  int c = (int)(i % C);
  size_t m = i / C;                  // m = ((b*H)+h)*W + w
  int wv = (int)(m % WW);
  int hv = (int)((m / WW) % HH);
  int bv = (int)(m / ((size_t)HH * WW));
  out[i] = in[(((size_t)bv * C + c) * HH + hv) * WW + wv];
}

__global__ __launch_bounds__(256)
void k_nhwc2nchw(const float* __restrict__ in, float* __restrict__ out, int C) {
  size_t i = (size_t)blockIdx.x * 256 + threadIdx.x;   // NCHW flat index
  size_t total = (size_t)MM * C;
  if (i >= total) return;
  int wv = (int)(i % WW);
  int hv = (int)((i / WW) % HH);
  int c  = (int)((i / ((size_t)HH * WW)) % C);
  int bv = (int)(i / ((size_t)C * HH * WW));
  size_t m = ((size_t)bv * HH + hv) * WW + wv;
  out[i] = in[m * C + c];
}

// =====================================================================
// Offset/mask conv (plain conv, NHWC input). aux[m][3K]:
//   j < 2K : raw offsets, channel j = 2*kk + (0=y,1=x)
//   j >= 2K: sigmoid(mask) for tap j-2K
// =====================================================================
__global__ __launch_bounds__(256)
void k_aux(const float* __restrict__ act, int C, int K, int kh, int kw, int pad,
           const float* __restrict__ offw, const float* __restrict__ offb,
           const float* __restrict__ mw,   const float* __restrict__ mb,
           float* __restrict__ aux) {
  int J = 3 * K;
  int tid = blockIdx.x * 256 + threadIdx.x;
  if (tid >= MM * J) return;
  int j = tid % J;
  int m = tid / J;
  int bv = m / (HH * WW);
  int hw = m % (HH * WW);
  int ho = hw / WW, wo = hw % WW;

  const float* wt; float bias; bool sig;
  if (j < 2 * K) { wt = offw + (size_t)j * C * K; bias = offb[j]; sig = false; }
  else { int kkm = j - 2 * K; wt = mw + (size_t)kkm * C * K; bias = mb[kkm]; sig = true; }

  float acc = bias;
  for (int ty = 0; ty < kh; ++ty) {
    int y = ho - pad + ty;
    if (y < 0 || y >= HH) continue;
    for (int tx = 0; tx < kw; ++tx) {
      int xp = wo - pad + tx;
      if (xp < 0 || xp >= WW) continue;
      const float* ar = act + ((size_t)((bv * HH + y) * WW + xp)) * C;
      const float* wr = wt + ty * kw + tx;          // per-channel stride K
      for (int c = 0; c < C; ++c) acc += ar[c] * wr[(size_t)c * K];
    }
  }
  aux[(size_t)m * J + j] = sig ? 1.f / (1.f + expf(-acc)) : acc;
}

// =====================================================================
// Weight prep: w[O][C][kh][kw] fp32 -> wbf[o][kk*C + c] bf16 (kk-major,
// matching the col layout so the GEMM needs no transpose)
// =====================================================================
__global__ __launch_bounds__(256)
void k_wprep(const float* __restrict__ w, __hip_bfloat16* __restrict__ wb,
             int O, int C, int K) {
  int tid = blockIdx.x * 256 + threadIdx.x;
  if (tid >= O * C * K) return;
  int o  = tid / (C * K);
  int rm = tid % (C * K);
  int c  = rm / K;
  int kk = rm % K;
  wb[(size_t)o * C * K + (size_t)kk * C + c] = __float2bfloat16(w[tid]);
}

// =====================================================================
// Modulated bilinear im2col: col[m][kk*C + c] bf16 (NHWC source, coalesced)
// One workgroup per (m, kk) tap; lanes sweep channels.
// =====================================================================
__global__ __launch_bounds__(128)
void k_sample(const float* __restrict__ act, const float* __restrict__ aux,
              __hip_bfloat16* __restrict__ col, int C, int K, int kw, int pad) {
  int mk = blockIdx.x;
  int kk = mk % K;
  int m  = mk / K;
  int bv = m / (HH * WW);
  int hw = m % (HH * WW);
  int ho = hw / WW, wo = hw % WW;

  const float* a = aux + (size_t)m * 3 * K;
  float offy = a[2 * kk], offx = a[2 * kk + 1], msk = a[2 * K + kk];
  int ky = kk / kw, kx = kk % kw;
  float gy = (float)(ho - pad + ky) + offy;
  float gx = (float)(wo - pad + kx) + offx;
  float y0f = floorf(gy), x0f = floorf(gx);
  int y0 = (int)y0f, x0 = (int)x0f;
  float wy = gy - y0f, wx = gx - x0f;

  bool vy0 = (y0 >= 0) && (y0 < HH);
  bool vy1 = (y0 + 1 >= 0) && (y0 + 1 < HH);
  bool vx0 = (x0 >= 0) && (x0 < WW);
  bool vx1 = (x0 + 1 >= 0) && (x0 + 1 < WW);
  float f00 = (vy0 && vx0) ? (1.f - wy) * (1.f - wx) * msk : 0.f;
  float f01 = (vy0 && vx1) ? (1.f - wy) * wx * msk : 0.f;
  float f10 = (vy1 && vx0) ? wy * (1.f - wx) * msk : 0.f;
  float f11 = (vy1 && vx1) ? wy * wx * msk : 0.f;

  int yc0 = min(max(y0, 0), HH - 1), yc1 = min(max(y0 + 1, 0), HH - 1);
  int xc0 = min(max(x0, 0), WW - 1), xc1 = min(max(x0 + 1, 0), WW - 1);
  const float* r00 = act + ((size_t)((bv * HH + yc0) * WW + xc0)) * C;
  const float* r01 = act + ((size_t)((bv * HH + yc0) * WW + xc1)) * C;
  const float* r10 = act + ((size_t)((bv * HH + yc1) * WW + xc0)) * C;
  const float* r11 = act + ((size_t)((bv * HH + yc1) * WW + xc1)) * C;

  __hip_bfloat16* dst = col + (size_t)m * K * C + (size_t)kk * C;
  for (int c = threadIdx.x; c < C; c += 128) {
    float v = f00 * r00[c] + f01 * r01[c] + f10 * r10[c] + f11 * r11[c];
    dst[c] = __float2bfloat16(v);
  }
}

// =====================================================================
// WMMA GEMM: out[m][n] = col[m][:] . wt[n][:] + bias[n] (+ res[m][n])
// bf16 inputs, f32 accumulate via v_wmma_f32_16x16x32_bf16.
// Tile 128x64x32, 8 waves (4x2), 32x32 per wave (2x2 WMMA accumulators).
// Tiles staged via async global->LDS DMA (ASYNCcnt), double-buffered.
// =====================================================================
#define TM 128
#define TN 64
#define TK 32
#define LDT 40   // padded LDS row stride (bf16 elems) = 80B: conflict-free, 16B aligned

// Async DMA: copy 16B from global to LDS, tracked by ASYNCcnt (gfx1250).
__device__ __forceinline__ void async_b128(unsigned ldsOff, const void* g) {
  asm volatile("global_load_async_to_lds_b128 %0, %1, off"
               :: "v"(ldsOff), "v"(g) : "memory");
}

__global__ __launch_bounds__(256)
void k_gemm(const __hip_bfloat16* __restrict__ colb,
            const __hip_bfloat16* __restrict__ wt,
            const float* __restrict__ bias,
            const float* __restrict__ res,
            float* __restrict__ out, int N, int KC) {
  __shared__ unsigned short As[2][TM * LDT];
  __shared__ unsigned short Bs[2][TN * LDT];

  const int m0 = blockIdx.x * TM;
  const int n0 = blockIdx.y * TN;
  const int tid  = threadIdx.x;
  const int wave = tid >> 5;
  const int lane = tid & 31;
  const int half = lane >> 4;   // wave32: lanes 0-15 / 16-31
  const int r    = lane & 15;
  const int wm = wave >> 1;     // 0..3 (M sub-tile)
  const int wn = wave & 1;      // 0..1 (N sub-tile)

  v8f acc[2][2];
  for (int ni = 0; ni < 2; ++ni) {
    float bv = bias[n0 + wn * 32 + ni * 16 + r];
    for (int mi = 0; mi < 2; ++mi)
      for (int v = 0; v < 8; ++v) acc[mi][ni][v] = bv;
  }

  const unsigned short* colu = (const unsigned short*)colb;
  const unsigned short* wtu  = (const unsigned short*)wt;
  const int arow = tid >> 1, aseg = tid & 1;   // 128 rows x 2 segs of 16 bf16
  const int brow = tid >> 2, bseg = tid & 3;   // 64 rows x 4 segs of 8 bf16

  // this thread's global source rows and LDS destination slots
  const unsigned short* aPtr = colu + (size_t)(m0 + arow) * KC + aseg * 16;
  const unsigned short* bPtr = wtu  + (size_t)(n0 + brow) * KC + bseg * 8;
  unsigned aOff[2], bOff[2];   // LDS byte offsets (generic-ptr low 32 bits)
  for (int s = 0; s < 2; ++s) {
    aOff[s] = (unsigned)(uintptr_t)&As[s][arow * LDT + aseg * 16];
    bOff[s] = (unsigned)(uintptr_t)&Bs[s][brow * LDT + bseg * 8];
  }

  auto issue = [&](int k0, int s) {   // 3 async b128 per thread per tile
    async_b128(aOff[s],      aPtr + k0);
    async_b128(aOff[s] + 16, aPtr + k0 + 8);
    async_b128(bOff[s],      bPtr + k0);
  };

  const int nt = KC / TK;
  issue(0, 0);   // prologue: stage tile 0

  for (int i = 0; i < nt; ++i) {
    const int buf = i & 1;
    if (i + 1 < nt) {
      issue((i + 1) * TK, buf ^ 1);                 // stage next tile
      asm volatile("s_wait_asynccnt 0x3" ::: "memory");  // retire tile i only
    } else {
      asm volatile("s_wait_asynccnt 0x0" ::: "memory");
    }
    __syncthreads();   // tile i visible to all waves

    // A fragment, 16-bit 16x32 layout: lane holds row M=r; chunks at
    // K = half*8 + 0..7 and K = 16 + half*8 + 0..7 (two b128 LDS reads).
    u32x4 afl[2], afh[2], bfl[2], bfh[2];
    for (int mi = 0; mi < 2; ++mi) {
      int row = wm * 32 + mi * 16 + r;
      afl[mi] = *(const u32x4*)&As[buf][row * LDT + half * 8];
      afh[mi] = *(const u32x4*)&As[buf][row * LDT + 16 + half * 8];
    }
    for (int ni = 0; ni < 2; ++ni) {
      int row = wn * 32 + ni * 16 + r;   // wt stored N-major => same chunking
      bfl[ni] = *(const u32x4*)&Bs[buf][row * LDT + half * 8];
      bfh[ni] = *(const u32x4*)&Bs[buf][row * LDT + 16 + half * 8];
    }
    struct F { u32x4 lo, hi; };
    for (int mi = 0; mi < 2; ++mi)
      for (int ni = 0; ni < 2; ++ni) {
        F fa{afl[mi], afh[mi]}, fb{bfl[ni], bfh[ni]};
        acc[mi][ni] = __builtin_amdgcn_wmma_f32_16x16x32_bf16(
            false, __builtin_bit_cast(v16bf, fa),
            false, __builtin_bit_cast(v16bf, fb),
            (short)0, acc[mi][ni], false, false);
      }
    __syncthreads();   // all waves done with buf before it is re-staged
  }

  // C/D layout: VGPR v -> row M = v + 8*half ; lane col N = r
  for (int mi = 0; mi < 2; ++mi)
    for (int ni = 0; ni < 2; ++ni)
      for (int v = 0; v < 8; ++v) {
        int m = m0 + wm * 32 + mi * 16 + half * 8 + v;
        int n = n0 + wn * 32 + ni * 16 + r;
        float o = acc[mi][ni][v];
        if (res) o += res[(size_t)m * N + n];
        out[(size_t)m * N + n] = o;
      }
}

// =====================================================================
// Training-mode BN: per-channel batch stats over M, then normalize+ReLU
// =====================================================================
__global__ __launch_bounds__(256)
void k_bnstats(const float* __restrict__ y, int N,
               float* __restrict__ mean, float* __restrict__ istd) {
  int n = blockIdx.x;
  float s = 0.f, q = 0.f;
  for (int m = threadIdx.x; m < MM; m += 256) {
    float v = y[(size_t)m * N + n];
    s += v; q += v * v;
  }
  __shared__ float ss[256], qq[256];
  ss[threadIdx.x] = s; qq[threadIdx.x] = q;
  __syncthreads();
  for (int st = 128; st > 0; st >>= 1) {
    if (threadIdx.x < st) {
      ss[threadIdx.x] += ss[threadIdx.x + st];
      qq[threadIdx.x] += qq[threadIdx.x + st];
    }
    __syncthreads();
  }
  if (threadIdx.x == 0) {
    float mu  = ss[0] / (float)MM;
    float var = qq[0] / (float)MM - mu * mu;
    mean[n] = mu;
    istd[n] = rsqrtf(var + EPSI);
  }
}

__global__ __launch_bounds__(256)
void k_bnapply(float* __restrict__ y, const float* __restrict__ mean,
               const float* __restrict__ istd, const float* __restrict__ gamma,
               const float* __restrict__ beta, int N) {
  size_t i = (size_t)blockIdx.x * 256 + threadIdx.x;
  size_t total = (size_t)MM * N;
  if (i >= total) return;
  int n = (int)(i % N);
  float v = (y[i] - mean[n]) * istd[n] * gamma[n] + beta[n];
  y[i] = v > 0.f ? v : 0.f;
}

// =====================================================================
// Host sequencing
// Input order (setup_inputs insertion order, recursive):
// 0:x  1-6:conv1_stride{w,b,ow,ob,mw,mb}  7-12:conv1  13-18:conv2
// 19-24:conv3  25-30:proj  31-36:proj_stride  37,38:bn_mid{g,b}  39,40:bn_out{g,b}
// =====================================================================
extern "C" void kernel_launch(void* const* d_in, const int* in_sizes, int n_in,
                              void* d_out, int out_size, void* d_ws, size_t ws_size,
                              hipStream_t stream) {
  auto P = [&](int i) { return (const float*)d_in[i]; };
  const float* x = P(0);

  char* ws = (char*)d_ws;
  size_t off = 0;
  auto carve = [&](size_t bytes) {
    void* p = ws + off;
    off = (off + bytes + 255) & ~(size_t)255;
    return p;
  };
  const size_t actBytes = (size_t)MM * 256 * sizeof(float);
  float* bufA = (float*)carve(actBytes);
  float* bufB = (float*)carve(actBytes);
  float* bufC = (float*)carve(actBytes);
  float* bufD = (float*)carve(actBytes);
  __hip_bfloat16* colB = (__hip_bfloat16*)carve((size_t)MM * 1152 * sizeof(__hip_bfloat16));
  float* auxB = (float*)carve((size_t)MM * 27 * sizeof(float));
  __hip_bfloat16* wbfB = (__hip_bfloat16*)carve((size_t)256 * 1152 * sizeof(__hip_bfloat16));
  float* meanB = (float*)carve(256 * sizeof(float));
  float* istdB = (float*)carve(256 * sizeof(float));

  auto deform = [&](const float* act, int C, int K, int kh, int kw, int pad,
                    int wi, float* outp, int N, const float* res) {
    int taux = MM * 3 * K;
    k_aux<<<(taux + 255) / 256, 256, 0, stream>>>(act, C, K, kh, kw, pad,
        P(wi + 2), P(wi + 3), P(wi + 4), P(wi + 5), auxB);
    int tw = N * C * K;
    k_wprep<<<(tw + 255) / 256, 256, 0, stream>>>(P(wi), wbfB, N, C, K);
    k_sample<<<MM * K, 128, 0, stream>>>(act, auxB, colB, C, K, kw, pad);
    dim3 g(MM / TM, N / TN);
    k_gemm<<<g, 256, 0, stream>>>(colB, wbfB, P(wi + 1), res, outp, N, C * K);
  };
  auto bn = [&](float* y, int N, const float* g, const float* b) {
    k_bnstats<<<N, 256, 0, stream>>>(y, N, meanB, istdB);
    size_t total = (size_t)MM * N;
    k_bnapply<<<(int)((total + 255) / 256), 256, 0, stream>>>(y, meanB, istdB, g, b, N);
  };

  // x -> NHWC
  k_nchw2nhwc<<<(MM * 128 + 255) / 256, 256, 0, stream>>>(x, bufA, 128);

  // ---- block 0 ----
  deform(bufA, 128, 1, 1, 1, 0, 1,  bufB, 128, nullptr);  bn(bufB, 128, P(37), P(38));
  deform(bufB, 128, 9, 3, 3, 1, 13, bufC, 128, nullptr);  bn(bufC, 128, P(37), P(38));
  deform(bufA, 128, 1, 1, 1, 0, 31, bufD, 256, nullptr);           // proj_stride identity
  deform(bufC, 128, 1, 1, 1, 0, 19, bufB, 256, bufD);     bn(bufB, 256, P(39), P(40));

  // ---- block 1 ---- (identity = bufB, 256 ch, no projection)
  deform(bufB, 256, 1, 1, 1, 0, 7,  bufA, 128, nullptr);  bn(bufA, 128, P(37), P(38));
  deform(bufA, 128, 9, 3, 3, 1, 13, bufC, 128, nullptr);  bn(bufC, 128, P(37), P(38));
  deform(bufC, 128, 1, 1, 1, 0, 19, bufD, 256, bufB);     bn(bufD, 256, P(39), P(40));

  // NHWC -> NCHW fp32 output
  k_nhwc2nchw<<<(MM * 256 + 255) / 256, 256, 0, stream>>>(bufD, (float*)d_out, 256);
}
